// DeepMinAttLSTM_76072460747247
// MI455X (gfx1250) — matrix operand order repaired and checked by reading, not compiled, not run on hardware
//
#include <hip/hip_runtime.h>
#include <stdint.h>

// ---------------- types ----------------
typedef __attribute__((ext_vector_type(16))) __bf16 v16bf;
typedef __attribute__((ext_vector_type(8)))  float  v8f;
typedef int v4i_t __attribute__((vector_size(16)));

union Frag {
    v16bf v;
    uint4 q[2];
    unsigned short h[16];
};

static __device__ __forceinline__ unsigned short f32_to_bf16(float f) {
    unsigned int u = __float_as_uint(f);
    unsigned int r = (u + 0x7FFFu + ((u >> 16) & 1u)) >> 16;
    return (unsigned short)r;
}

// -------- CDNA5 async memory->LDS path (guarded; falls back to batched ld/st) --------
#if __has_builtin(__builtin_amdgcn_global_load_async_to_lds_b128)
#define HAVE_ASYNC_LDS 1
static __device__ __forceinline__ void async_cp16(const void* g, void* l) {
    __builtin_amdgcn_global_load_async_to_lds_b128(
        (__attribute__((address_space(1))) v4i_t*)g,
        (__attribute__((address_space(3)))  v4i_t*)l, 0, 0);
}
#if __has_builtin(__builtin_amdgcn_s_wait_asynccnt)
#define WAIT_ASYNC(n) __builtin_amdgcn_s_wait_asynccnt(n)
#else
#define WAIT_ASYNC(n) asm volatile("s_wait_asynccnt %0" ::"i"(n) : "memory")
#endif
#else
#define HAVE_ASYNC_LDS 0
#define WAIT_ASYNC(n)
#endif

// ---------------- fp32 -> bf16 convert ----------------
__global__ void k_cvt_bf16(const float* __restrict__ src,
                           unsigned short* __restrict__ dst, long long n) {
    long long i = (long long)blockIdx.x * blockDim.x + threadIdx.x;
    long long stride = (long long)gridDim.x * blockDim.x;
    for (; i < n; i += stride) dst[i] = f32_to_bf16(src[i]);
}

// ---------------- V transpose-convert: (B,S,H) f32 -> (B,NH,dh,S) bf16 ----------------
__global__ void k_vt_cvt(const float* __restrict__ v, unsigned short* __restrict__ vT,
                         int Bsz, int S, int H, int NH) {
    long long n = (long long)Bsz * S * H;
    int dh = H / NH;
    long long i = (long long)blockIdx.x * blockDim.x + threadIdx.x;
    long long stride = (long long)gridDim.x * blockDim.x;
    for (; i < n; i += stride) {
        int hcol = (int)(i % H);
        long long t = i / H;
        int s = (int)(t % S);
        int b = (int)(t / S);
        int head = hcol / dh, d = hcol % dh;
        vT[(((size_t)b * NH + head) * dh + d) * S + s] = f32_to_bf16(v[i]);
    }
}

// ---------------- WMMA bf16 GEMM: C[M,N] = A[M,K] * W[N,K]^T + bias[N] ----------------
// Block: 256 thr = 8 waves. Tile 128x128, K-step 32. Wave = 32(M) x 64(N) = 2x4 wmma tiles.
// Double-buffered LDS staged via GLOBAL_LOAD_ASYNC_TO_LDS_B128 (ASYNCcnt pipelined).
#define TM 128
#define TN 128
#define TK 32
#define LDP 40   // padded K stride (16B-aligned, reduces bank conflicts)

static __device__ __forceinline__ void stage_tile(
    const unsigned short* __restrict__ A, const unsigned short* __restrict__ W,
    unsigned short (*__restrict__ As)[LDP], unsigned short (*__restrict__ Bs)[LDP],
    int bm, int bn, int K, int kk, int tid) {
#if HAVE_ASYNC_LDS
#pragma unroll
    for (int s = 0; s < 2; ++s) {
        int slot = tid + s * 256;          // 0..511
        int row  = slot >> 2;              // 0..127
        int c16  = (slot & 3) * 8;         // 0,8,16,24
        async_cp16(&A[(size_t)(bm + row) * K + kk + c16], &As[row][c16]);
        async_cp16(&W[(size_t)(bn + row) * K + kk + c16], &Bs[row][c16]);
    }
#else
    uint4 va[2], vb[2];
#pragma unroll
    for (int s = 0; s < 2; ++s) {
        int slot = tid + s * 256;
        int row  = slot >> 2;
        int c16  = (slot & 3) * 8;
        va[s] = *(const uint4*)&A[(size_t)(bm + row) * K + kk + c16];
        vb[s] = *(const uint4*)&W[(size_t)(bn + row) * K + kk + c16];
    }
#pragma unroll
    for (int s = 0; s < 2; ++s) {
        int slot = tid + s * 256;
        int row  = slot >> 2;
        int c16  = (slot & 3) * 8;
        *(uint4*)&As[row][c16] = va[s];
        *(uint4*)&Bs[row][c16] = vb[s];
    }
#endif
}

__global__ __launch_bounds__(256)
void k_gemm_bf16(const unsigned short* __restrict__ A,
                 const unsigned short* __restrict__ W,
                 const float* __restrict__ bias,
                 float* __restrict__ C, int M, int N, int K) {
    __shared__ unsigned short As[2][TM][LDP];   // 2 x 10 KB
    __shared__ unsigned short Bs[2][TN][LDP];   // 2 x 10 KB

    const int tid  = threadIdx.x;
    const int lane = tid & 31;
    const int wid  = tid >> 5;
    const int wm   = wid >> 1;     // 0..3 -> +32*wm in M
    const int wn   = wid & 1;      // 0..1 -> +64*wn in N
    const int l15  = lane & 15;
    const int kg   = lane >> 4;

    const int nTilesN = N / TN;
    const int bm = (blockIdx.x / nTilesN) * TM;
    const int bn = (blockIdx.x % nTilesN) * TN;

    v8f acc[2][4];
#pragma unroll
    for (int i = 0; i < 2; ++i)
#pragma unroll
        for (int j = 0; j < 4; ++j) acc[i][j] = (v8f){0.f,0.f,0.f,0.f,0.f,0.f,0.f,0.f};

    const int nk = K / TK;
    stage_tile(A, W, As[0], Bs[0], bm, bn, K, 0, tid);

    for (int kt = 0; kt < nk; ++kt) {
        const int cur = kt & 1;
        if (kt + 1 < nk) {
            stage_tile(A, W, As[cur ^ 1], Bs[cur ^ 1], bm, bn, K, (kt + 1) * TK, tid);
            // hint the tile after next into L2 (global_prefetch_b8)
            if (kt + 2 < nk) {
                __builtin_prefetch(&A[(size_t)(bm + (tid >> 1)) * K + (kt + 2) * TK], 0, 3);
                __builtin_prefetch(&W[(size_t)(bn + (tid >> 1)) * K + (kt + 2) * TK], 0, 3);
            }
            WAIT_ASYNC(4);      // in-order ASYNCcnt: tile kt's 4 copies complete
        } else {
            WAIT_ASYNC(0);
        }
        __syncthreads();        // all waves' tile-kt data visible in LDS

        // A fragments (16x32, MxK): lane row = l15, 8 contig K at kg*8 and 16+kg*8
        Frag a[2];
#pragma unroll
        for (int i = 0; i < 2; ++i) {
            int r = wm * 32 + i * 16 + l15;
            a[i].q[0] = *(const uint4*)&As[cur][r][kg * 8];
            a[i].q[1] = *(const uint4*)&As[cur][r][16 + kg * 8];
        }
        // B fragments (32x16, KxN): lane col = l15, 8 contig K at kg*8 and 16+kg*8
        Frag b[4];
#pragma unroll
        for (int j = 0; j < 4; ++j) {
            int c = wn * 64 + j * 16 + l15;
            b[j].q[0] = *(const uint4*)&Bs[cur][c][kg * 8];
            b[j].q[1] = *(const uint4*)&Bs[cur][c][16 + kg * 8];
        }
#pragma unroll
        for (int i = 0; i < 2; ++i)
#pragma unroll
            for (int j = 0; j < 4; ++j)
                acc[i][j] = __builtin_amdgcn_wmma_f32_16x16x32_bf16(
                    false, a[i].v, false, b[j].v, (short)0, acc[i][j], false, false);
        __syncthreads();        // all reads of As/Bs[cur] done before it is re-staged
    }

    // epilogue: C layout row = r + 8*kg, col = l15 per 16x16 subtile
#pragma unroll
    for (int i = 0; i < 2; ++i)
#pragma unroll
        for (int j = 0; j < 4; ++j) {
            int n = bn + wn * 64 + j * 16 + l15;
            float bv = bias ? bias[n] : 0.f;
#pragma unroll
            for (int r = 0; r < 8; ++r) {
                int m = bm + wm * 32 + i * 16 + r + kg * 8;
                C[(size_t)m * N + n] = acc[i][j][r] + bv;
            }
        }
}

// ---------------- minLSTM gate + sequential scan ----------------
__global__ void k_gate_scan(const float* __restrict__ fpre, const float* __restrict__ ipre,
                            const float* __restrict__ hpre, float* __restrict__ hout,
                            unsigned short* __restrict__ hbf, int Bsz, int S, int H) {
    int idx = blockIdx.x * blockDim.x + threadIdx.x;
    if (idx >= Bsz * H) return;
    int b = idx / H, hc = idx % H;
    size_t base = (size_t)b * S * H + hc;
    float h = 0.f;
    for (int t = 0; t < S; ++t) {
        size_t off = base + (size_t)t * H;
        float f = 1.f / (1.f + __expf(-fpre[off]));
        float i = 1.f / (1.f + __expf(-ipre[off]));
        float inv = 1.f / (f + i);
        h = f * inv * h + i * inv * hpre[off];
        hout[off] = h;
        hbf[off]  = f32_to_bf16(h);
    }
}

// ---------------- flash attention, bf16 WMMA, online softmax ----------------
#define DH 128
#define CH 32

__global__ __launch_bounds__(128)
void k_attn(const unsigned short* __restrict__ Q, const unsigned short* __restrict__ Km,
            const unsigned short* __restrict__ VT, unsigned short* __restrict__ O,
            int Bsz, int NH, int S) {
    __shared__ unsigned short Plds[4][16][CH + 8];   // per-wave P staging (C->A relayout)

    const int tid  = threadIdx.x;
    const int lane = tid & 31;
    const int widx = tid >> 5;
    const int l15  = lane & 15;
    const int kg   = lane >> 4;
    const int H    = NH * DH;

    const int qtiles = S / 16;
    const int w    = blockIdx.x * 4 + widx;
    const int qt   = w % qtiles;
    const int head = (w / qtiles) % NH;
    const int b    = w / (qtiles * NH);

    Frag qf[4];
    {
        const unsigned short* qrow = Q + ((size_t)b * S + qt * 16 + l15) * H + head * DH;
#pragma unroll
        for (int ks = 0; ks < 4; ++ks) {
            qf[ks].q[0] = *(const uint4*)&qrow[ks * 32 + kg * 8];
            qf[ks].q[1] = *(const uint4*)&qrow[ks * 32 + 16 + kg * 8];
        }
    }

    v8f oacc[8];
#pragma unroll
    for (int d = 0; d < 8; ++d) oacc[d] = (v8f){0.f,0.f,0.f,0.f,0.f,0.f,0.f,0.f};
    float runmax[8], runsum[8];
#pragma unroll
    for (int r = 0; r < 8; ++r) { runmax[r] = -1e30f; runsum[r] = 0.f; }

    const float scale = 0.08838834764831845f;   // 1/sqrt(128)

    for (int c0 = 0; c0 < S; c0 += CH) {
        v8f s0 = (v8f){0.f,0.f,0.f,0.f,0.f,0.f,0.f,0.f};
        v8f s1 = s0;
#pragma unroll
        for (int ns = 0; ns < 2; ++ns) {
            const unsigned short* krow =
                Km + ((size_t)b * S + c0 + ns * 16 + l15) * H + head * DH;
#pragma unroll
            for (int ks = 0; ks < 4; ++ks) {
                Frag kf;
                kf.q[0] = *(const uint4*)&krow[ks * 32 + kg * 8];
                kf.q[1] = *(const uint4*)&krow[ks * 32 + 16 + kg * 8];
                if (ns == 0)
                    s0 = __builtin_amdgcn_wmma_f32_16x16x32_bf16(false, qf[ks].v, false, kf.v,
                                                                 (short)0, s0, false, false);
                else
                    s1 = __builtin_amdgcn_wmma_f32_16x16x32_bf16(false, qf[ks].v, false, kf.v,
                                                                 (short)0, s1, false, false);
            }
        }

#pragma unroll
        for (int r = 0; r < 8; ++r) {
            float a0 = s0[r] * scale, a1 = s1[r] * scale;
            float m = fmaxf(a0, a1);
#pragma unroll
            for (int d = 1; d < 16; d <<= 1) m = fmaxf(m, __shfl_xor(m, d, 32));
            float nm   = fmaxf(runmax[r], m);
            float corr = __expf(runmax[r] - nm);
            runmax[r]  = nm;
            float p0 = __expf(a0 - nm);
            float p1 = __expf(a1 - nm);
            float rs = p0 + p1;
#pragma unroll
            for (int d = 1; d < 16; d <<= 1) rs += __shfl_xor(rs, d, 32);
            runsum[r] = runsum[r] * corr + rs;
#pragma unroll
            for (int db = 0; db < 8; ++db) oacc[db][r] *= corr;
            s0[r] = p0; s1[r] = p1;
        }

#pragma unroll
        for (int r = 0; r < 8; ++r) {
            int m = r + kg * 8;
            Plds[widx][m][l15]      = f32_to_bf16(s0[r]);
            Plds[widx][m][16 + l15] = f32_to_bf16(s1[r]);
        }
        __syncthreads();     // uniform trip count across waves
        Frag pf;
        pf.q[0] = *(const uint4*)&Plds[widx][l15][kg * 8];
        pf.q[1] = *(const uint4*)&Plds[widx][l15][16 + kg * 8];

#pragma unroll
        for (int db = 0; db < 8; ++db) {
            const unsigned short* vp =
                VT + (((size_t)b * NH + head) * DH + db * 16 + l15) * S + c0 + kg * 8;
            Frag vf;
            vf.q[0] = *(const uint4*)&vp[0];
            vf.q[1] = *(const uint4*)&vp[16];
            oacc[db] = __builtin_amdgcn_wmma_f32_16x16x32_bf16(false, pf.v, false, vf.v,
                                                               (short)0, oacc[db], false, false);
        }
        __syncthreads();
    }

#pragma unroll
    for (int db = 0; db < 8; ++db) {
#pragma unroll
        for (int r = 0; r < 8; ++r) {
            int srow = qt * 16 + r + kg * 8;
            int d    = db * 16 + l15;
            O[((size_t)b * S + srow) * H + head * DH + d] =
                f32_to_bf16(oacc[db][r] / runsum[r]);
        }
    }
}

// ---------------- final FC on last token (tiny: 8 MFLOP) ----------------
__global__ void k_fc(const float* __restrict__ attn, const float* __restrict__ res,
                     const float* __restrict__ fcw, const float* __restrict__ fcb,
                     float* __restrict__ out, int Bsz, int S, int H, int O) {
    int idx = blockIdx.x * blockDim.x + threadIdx.x;
    if (idx >= Bsz * O) return;
    int b = idx / O, o = idx % O;
    const float* a = attn + ((size_t)b * S + (S - 1)) * H;
    const float* r = res  + ((size_t)b * S + (S - 1)) * H;
    const float* wp = fcw + (size_t)o * H;
    float acc = fcb[o];
    for (int h = 0; h < H; ++h) acc = fmaf(a[h] + r[h], wp[h], acc);
    out[idx] = acc;
}

// ---------------- host launcher ----------------
extern "C" void kernel_launch(void* const* d_in, const int* in_sizes, int n_in,
                              void* d_out, int out_size, void* d_ws, size_t ws_size,
                              hipStream_t stream) {
    const float* x         = (const float*)d_in[0];
    const float* Wf        = (const float*)d_in[1];
    const float* bfv       = (const float*)d_in[2];
    const float* Wi        = (const float*)d_in[3];
    const float* biv       = (const float*)d_in[4];
    const float* Wh        = (const float*)d_in[5];
    const float* bhv       = (const float*)d_in[6];
    const float* in_proj_w = (const float*)d_in[7];
    const float* in_proj_b = (const float*)d_in[8];
    const float* out_w     = (const float*)d_in[9];
    const float* out_b     = (const float*)d_in[10];
    const float* fc_w      = (const float*)d_in[11];
    const float* fc_b      = (const float*)d_in[12];

    const int Bsz = 16, S = 1024, H = 1024, O = 256, L = 4, NH = 8;
    const int M = Bsz * S;                       // 16384
    const long long BSH = (long long)M * H;      // 16.7M elems

    char* ws = (char*)d_ws;
    unsigned short* xbf = (unsigned short*)(ws + (size_t)0);          // 32 MB
    float* g0  = (float*)(ws + ((size_t)32  << 20));                  // 64 MB
    float* g1  = (float*)(ws + ((size_t)96  << 20));                  // 64 MB
    float* g2  = (float*)(ws + ((size_t)160 << 20));                  // 64 MB
    float* hres = (float*)(ws + ((size_t)224 << 20));                 // 64 MB
    unsigned short* qbf = (unsigned short*)(ws + ((size_t)288 << 20));// 32 MB
    unsigned short* kbf = (unsigned short*)(ws + ((size_t)320 << 20));// 32 MB
    unsigned short* vT  = (unsigned short*)(ws + ((size_t)352 << 20));// 32 MB
    unsigned short* obf = (unsigned short*)(ws + ((size_t)384 << 20));// 32 MB
    unsigned short* wbf = (unsigned short*)(ws + ((size_t)416 << 20));// 6 MB staging

    dim3 gemmGrid((M / TM) * (H / TN));          // 128 * 8 = 1024 blocks

    k_cvt_bf16<<<4096, 256, 0, stream>>>(x, xbf, BSH);

    const long long HH = (long long)H * H;
    for (int l = 0; l < L; ++l) {
        size_t wo = (size_t)l * H * H;
        k_cvt_bf16<<<1024, 256, 0, stream>>>(Wf + wo, wbf + 0 * HH, HH);
        k_cvt_bf16<<<1024, 256, 0, stream>>>(Wi + wo, wbf + 1 * HH, HH);
        k_cvt_bf16<<<1024, 256, 0, stream>>>(Wh + wo, wbf + 2 * HH, HH);
        k_gemm_bf16<<<gemmGrid, 256, 0, stream>>>(xbf, wbf + 0 * HH, bfv + l * H, g0, M, H, H);
        k_gemm_bf16<<<gemmGrid, 256, 0, stream>>>(xbf, wbf + 1 * HH, biv + l * H, g1, M, H, H);
        k_gemm_bf16<<<gemmGrid, 256, 0, stream>>>(xbf, wbf + 2 * HH, bhv + l * H, g2, M, H, H);
        k_gate_scan<<<(Bsz * H + 255) / 256, 256, 0, stream>>>(g0, g1, g2, hres, xbf, Bsz, S, H);
    }

    k_cvt_bf16<<<3072, 256, 0, stream>>>(in_proj_w, wbf, 3 * HH);
    k_gemm_bf16<<<gemmGrid, 256, 0, stream>>>(xbf, wbf + 0 * HH, in_proj_b + 0 * H, g0, M, H, H);
    k_gemm_bf16<<<gemmGrid, 256, 0, stream>>>(xbf, wbf + 1 * HH, in_proj_b + 1 * H, g1, M, H, H);
    k_gemm_bf16<<<gemmGrid, 256, 0, stream>>>(xbf, wbf + 2 * HH, in_proj_b + 2 * H, g2, M, H, H);
    k_cvt_bf16<<<4096, 256, 0, stream>>>(g0, qbf, BSH);
    k_cvt_bf16<<<4096, 256, 0, stream>>>(g1, kbf, BSH);
    k_vt_cvt<<<4096, 256, 0, stream>>>(g2, vT, Bsz, S, H, NH);

    int nwaves = Bsz * NH * (S / 16);            // 8192 -> 2048 blocks of 4 waves
    k_attn<<<nwaves / 4, 128, 0, stream>>>(qbf, kbf, vT, obf, Bsz, NH, S);

    k_cvt_bf16<<<1024, 256, 0, stream>>>(out_w, wbf, HH);
    k_gemm_bf16<<<gemmGrid, 256, 0, stream>>>(obf, wbf, out_b, g0, M, H, H);
    k_fc<<<(Bsz * O + 255) / 256, 256, 0, stream>>>(g0, hres, fc_w, fc_b,
                                                    (float*)d_out, Bsz, S, H, O);
}
